// ChannelGroupVectorQuantizer_47218870452254
// MI455X (gfx1250) — matrix operand order repaired and checked by reading, not compiled, not run on hardware
//
#include <hip/hip_runtime.h>

// ---------------------------------------------------------------------------
// ChannelGroupVectorQuantizer for MI455X (gfx1250, wave32, WMMA)
//
// Per group g: scores S[pos,k] = sum_c x[pos,c]*e[k,c] via
// V_WMMA_F32_16X16X32_F16 (A = 16 pos x 32 chan, B = 32 chan x 16 codes).
// f16 hi/lo split (3 WMMAs) gives ~f32-accurate xe so argmin matches the
// f32 reference. dist argmin == argmin_k (e2[k] - 2*xe[k]).
// Round 2: interleaved dual accumulator chains (fills WMMA->VALU hazard
// slots) + double-buffered B-fragment software pipeline (hides LDS latency).
// Outputs: [recon | zq | idx(int32)] concatenated flat.
// ---------------------------------------------------------------------------

typedef __attribute__((ext_vector_type(16))) _Float16 v16h;
typedef __attribute__((ext_vector_type(8)))  float    v8f;

#define NCHW   (16 * 256 * 1024)        // 4,194,304
#define CB_ROW 20                        // dwords per codebook LDS row (16 data + 4 pad)
#define XS_ROW 36                        // dwords per x-stage row (32 data + 4 pad)

// LDS layout (dword offsets)
#define OFF_CBHI 0
#define OFF_CBLO (OFF_CBHI + 512 * CB_ROW)           // 10240
#define OFF_E2   (OFF_CBLO + 512 * CB_ROW)           // 20480
#define OFF_XS   (OFF_E2 + 512)                      // 20992
#define OFF_KB   (OFF_XS + 8 * 16 * XS_ROW)          // 25600
#define LDS_DW   (OFF_KB + 8 * 32)                   // 25856 dwords = 103,424 B

union BFrag { v16h h; struct { uint4 a, b; } q; };

__global__ void ChannelGroupVectorQuantizer_47218870452254_kernel(
    const float* __restrict__ feather,   // (16,256,32,32) f32
    const float* __restrict__ cb,        // (8,512,32) f32
    float* __restrict__ out)             // [recon | zq | idx]
{
    extern __shared__ __align__(16) unsigned int lds[];
    unsigned int* cbhi = lds + OFF_CBHI;
    unsigned int* cblo = lds + OFF_CBLO;
    float*        e2s  = (float*)(lds + OFF_E2);
    float*        xs   = (float*)(lds + OFF_XS);
    int*          kb   = (int*)(lds + OFF_KB);

    const int tid      = threadIdx.x;
    const int b        = blockIdx.x;     // 512 blocks
    const int n        = b >> 5;         // / (G * 4 chunks)
    const int g        = (b >> 2) & 7;
    const int posBase0 = (b & 3) << 8;   // 256-position chunk

    // ---- stage codebook group g: scaled (x256) f16 hi/lo rows + e2*256 ----
    for (int k = tid; k < 512; k += 256) {
        const float4* row = (const float4*)(cb + ((size_t)(g * 512 + k) << 5));
        float fr[32];
        float ss = 0.f;
        #pragma unroll
        for (int j = 0; j < 8; ++j) {
            float4 v = row[j];
            fr[4 * j + 0] = v.x; fr[4 * j + 1] = v.y;
            fr[4 * j + 2] = v.z; fr[4 * j + 3] = v.w;
            ss += v.x * v.x + v.y * v.y + v.z * v.z + v.w * v.w;
        }
        e2s[k] = ss * 256.0f;            // same x256 scale as xe -> monotone
        unsigned int* dh = cbhi + k * CB_ROW;
        unsigned int* dl = cblo + k * CB_ROW;
        #pragma unroll
        for (int j = 0; j < 16; ++j) {
            union { _Float16 h[2]; unsigned int u; } ph, pl;
            #pragma unroll
            for (int s = 0; s < 2; ++s) {
                float f = fr[2 * j + s] * 256.0f;   // keep lo-part normal in f16
                _Float16 hi = (_Float16)f;
                ph.h[s] = hi;
                pl.h[s] = (_Float16)(f - (float)hi);
            }
            dh[j] = ph.u;
            dl[j] = pl.u;
        }
    }
    __syncthreads();

    const int w       = tid >> 5;
    const int lane    = tid & 31;
    const int lane16  = lane & 15;
    const int half    = lane >> 4;
    const int posWave = posBase0 + w * 32;          // 2 tiles of 16 positions

    const size_t chanRow = ((size_t)(n * 256 + g * 32 + lane)) << 10;

    float4 xv[2][4];                                 // keep x in regs for recon
    v16h   ahi[2], alo[2];
    float* xsw = xs + w * 16 * XS_ROW;               // per-wave transpose buffer

    // ---- load + transpose x tiles, build A fragments (hi/lo f16) ----
    #pragma unroll
    for (int p = 0; p < 2; ++p) {
        const float4* src = (const float4*)(feather + chanRow + posWave + p * 16);
        #pragma unroll
        for (int q = 0; q < 4; ++q) xv[p][q] = src[q];
        // write [pos][chan] (lane = channel): conflict-free dword stores
        #pragma unroll
        for (int q = 0; q < 4; ++q) {
            xsw[(4 * q + 0) * XS_ROW + lane] = xv[p][q].x;
            xsw[(4 * q + 1) * XS_ROW + lane] = xv[p][q].y;
            xsw[(4 * q + 2) * XS_ROW + lane] = xv[p][q].z;
            xsw[(4 * q + 3) * XS_ROW + lane] = xv[p][q].w;
        }
        __builtin_amdgcn_wave_barrier();  // wave-local; DS is in-order per wave
        // A layout (16-bit 16x32): lanes<16 -> K {0..7,16..23}; lanes>=16 -> {8..15,24..31}
        const int r = half ? 8 : 0;
        const float4* rp  = (const float4*)(xsw + lane16 * XS_ROW + r);
        const float4* rp2 = (const float4*)(xsw + lane16 * XS_ROW + r + 16);
        float fa[16];
        float4 a0 = rp[0], a1 = rp[1], a2 = rp2[0], a3 = rp2[1];
        fa[0]=a0.x; fa[1]=a0.y; fa[2]=a0.z; fa[3]=a0.w;
        fa[4]=a1.x; fa[5]=a1.y; fa[6]=a1.z; fa[7]=a1.w;
        fa[8]=a2.x; fa[9]=a2.y; fa[10]=a2.z; fa[11]=a2.w;
        fa[12]=a3.x; fa[13]=a3.y; fa[14]=a3.z; fa[15]=a3.w;
        #pragma unroll
        for (int i = 0; i < 16; ++i) {
            _Float16 h = (_Float16)fa[i];
            ahi[p][i] = h;
            alo[p][i] = (_Float16)(fa[i] - (float)h);
        }
        __builtin_amdgcn_wave_barrier();  // before p=1 reuses xsw
    }

    // ---- scan 32 code tiles of 16; per-lane running (min, argmin) ----
    float best[2][8];
    int   bidx[2][8];
    #pragma unroll
    for (int p = 0; p < 2; ++p)
        #pragma unroll
        for (int r = 0; r < 8; ++r) { best[p][r] = 3.402823466e+38f; bidx[p][r] = 0; }

    // double-buffered B fragments + e2 (software pipeline, depth 2)
    BFrag Bh[2], Bl[2];
    float e2k[2];
    {   // prologue: tile 0 -> buffer 0
        const unsigned int* bh = cbhi + lane16 * CB_ROW + half * 4;
        const unsigned int* bl = cblo + lane16 * CB_ROW + half * 4;
        Bh[0].q.a = *(const uint4*)(bh);  Bh[0].q.b = *(const uint4*)(bh + 8);
        Bl[0].q.a = *(const uint4*)(bl);  Bl[0].q.b = *(const uint4*)(bl + 8);
        e2k[0] = e2s[lane16];
    }

    #pragma unroll 2
    for (int t = 0; t < 32; ++t) {
        const int buf = t & 1;
        if (t + 1 < 32) {                // prefetch next tile while computing
            const int kn = ((t + 1) << 4) | lane16;
            const unsigned int* bh = cbhi + kn * CB_ROW + half * 4;
            const unsigned int* bl = cblo + kn * CB_ROW + half * 4;
            Bh[buf ^ 1].q.a = *(const uint4*)(bh);
            Bh[buf ^ 1].q.b = *(const uint4*)(bh + 8);
            Bl[buf ^ 1].q.a = *(const uint4*)(bl);
            Bl[buf ^ 1].q.b = *(const uint4*)(bl + 8);
            e2k[buf ^ 1] = e2s[kn];
        }
        const int kcode = (t << 4) | lane16;

        // two independent accumulation chains, interleaved so each chain's
        // WMMA->VALU / D->A hazard windows are filled by the other chain
        v8f acc0 = {}, acc1 = {};
        acc0 = __builtin_amdgcn_wmma_f32_16x16x32_f16(false, alo[0], false, Bh[buf].h,
                                                      (short)0, acc0, false, false);
        acc1 = __builtin_amdgcn_wmma_f32_16x16x32_f16(false, alo[1], false, Bh[buf].h,
                                                      (short)0, acc1, false, false);
        acc0 = __builtin_amdgcn_wmma_f32_16x16x32_f16(false, ahi[0], false, Bl[buf].h,
                                                      (short)0, acc0, false, false);
        acc1 = __builtin_amdgcn_wmma_f32_16x16x32_f16(false, ahi[1], false, Bl[buf].h,
                                                      (short)0, acc1, false, false);
        acc0 = __builtin_amdgcn_wmma_f32_16x16x32_f16(false, ahi[0], false, Bh[buf].h,
                                                      (short)0, acc0, false, false);
        acc1 = __builtin_amdgcn_wmma_f32_16x16x32_f16(false, ahi[1], false, Bh[buf].h,
                                                      (short)0, acc1, false, false);
        #pragma unroll
        for (int r = 0; r < 8; ++r) {
            float s0 = fmaf(acc0[r], -2.0f, e2k[buf]);   // scaled (e2 - 2*xe)
            if (s0 < best[0][r]) { best[0][r] = s0; bidx[0][r] = kcode; }
            float s1 = fmaf(acc1[r], -2.0f, e2k[buf]);
            if (s1 < best[1][r]) { best[1][r] = s1; bidx[1][r] = kcode; }
        }
    }

    // ---- argmin across the 16 columns (lanes within each half) ----
    #pragma unroll
    for (int p = 0; p < 2; ++p) {
        #pragma unroll
        for (int r = 0; r < 8; ++r) {
            float v = best[p][r];
            int   kk = bidx[p][r];
            #pragma unroll
            for (int m = 8; m >= 1; m >>= 1) {       // masks < 16: stay in half
                float ov = __shfl_xor(v, m, 32);
                int   ok = __shfl_xor(kk, m, 32);
                if (ov < v || (ov == v && ok < kk)) { v = ov; kk = ok; }
            }
            bidx[p][r] = kk;
        }
    }

    // lanes 0 / 16 own rows 0..7 / 8..15 of D -> publish idx + kbuf
    int* idxOut = ((int*)out) + (size_t)2 * NCHW;
    if (lane16 == 0) {
        const size_t idxBase = ((size_t)(n * 8 + g)) << 10;
        #pragma unroll
        for (int p = 0; p < 2; ++p) {
            #pragma unroll
            for (int r = 0; r < 8; ++r) {
                const int row = half * 8 + r;
                kb[w * 32 + p * 16 + row] = bidx[p][r];
                idxOut[idxBase + posWave + p * 16 + row] = bidx[p][r];
            }
        }
    }
    __builtin_amdgcn_wave_barrier();                 // wave-local kbuf handoff

    // ---- gather zq rows (f32 codebook via L2), emit recon = x + (zq - x) ----
    const float* cbg = cb + ((size_t)g << 14);       // g*512*32
    #pragma unroll
    for (int p = 0; p < 2; ++p) {
        float vals[16];
        #pragma unroll
        for (int i = 0; i < 16; ++i) {
            const int kk = kb[w * 32 + p * 16 + i];
            vals[i] = cbg[((size_t)kk << 5) + lane]; // lane = channel c
        }
        float4* zqDst = (float4*)(out + NCHW + chanRow + posWave + p * 16);
        float4* rcDst = (float4*)(out + chanRow + posWave + p * 16);
        #pragma unroll
        for (int q = 0; q < 4; ++q) {
            float4 z;
            z.x = vals[4 * q + 0]; z.y = vals[4 * q + 1];
            z.z = vals[4 * q + 2]; z.w = vals[4 * q + 3];
            float4 x = xv[p][q];
            float4 rc;                               // bit-match f + (z - f)
            rc.x = x.x + (z.x - x.x); rc.y = x.y + (z.y - x.y);
            rc.z = x.z + (z.z - x.z); rc.w = x.w + (z.w - x.w);
            zqDst[q] = z;
            rcDst[q] = rc;
        }
    }
}

extern "C" void kernel_launch(void* const* d_in, const int* in_sizes, int n_in,
                              void* d_out, int out_size, void* d_ws, size_t ws_size,
                              hipStream_t stream) {
    (void)in_sizes; (void)n_in; (void)out_size; (void)d_ws; (void)ws_size;
    const float* feather = (const float*)d_in[0];
    const float* cb      = (const float*)d_in[1];
    float*       out     = (float*)d_out;

    dim3 grid(512);          // N(16) * G(8) * 4 position-chunks of 256
    dim3 block(256);         // 8 wave32 waves per block
    size_t shmem = (size_t)LDS_DW * 4;   // 103,424 B dynamic LDS (<320KB/WGP)
    hipLaunchKernelGGL(ChannelGroupVectorQuantizer_47218870452254_kernel,
                       grid, block, shmem, stream, feather, cb, out);
}
